// MLP_22883585753836
// MI455X (gfx1250) — matrix operand order repaired and checked by reading, not compiled, not run on hardware
//
#include <hip/hip_runtime.h>
#include <hip/hip_bf16.h>

typedef __attribute__((ext_vector_type(16))) _Float16 v16h;
typedef __attribute__((ext_vector_type(8)))  _Float16 v8h;
typedef __attribute__((ext_vector_type(8)))  float    v8f;

union AV { v16h v; v8h h[2]; };

#define EDIM   64
#define E3     192
#define E2     128
#define VDIM   32000
#define TLEN   1024
#define NTOK   4096          // B*T = 4*1024
#define MTILES 256           // NTOK/16
#define NTILES 2000          // VDIM/16
#define NT_PER_WAVE 8
#define NGROUPS (NTILES / NT_PER_WAVE)   // 250

// ---------------------------------------------------------------------------
// Stage 0: pack W2 (f32 [128][32000] row-major) into f16 WMMA B-fragments.
// Dest layout: half index = ((ntile*4 + c)*32 + lane)*16 + s
//   lane < 16 : element (K = c*32 + (s<8 ? s : s+8),     N = ntile*16 + lane)
//   lane >= 16: element (K = c*32 + 8 + (s<8 ? s : s+8), N = ntile*16 + lane - 16)
// Each consuming lane then reads its 16 halves as one contiguous 32B v16h.
// ---------------------------------------------------------------------------
__global__ void pack_w2_f16(const float* __restrict__ W2,
                            _Float16* __restrict__ w2h) {
    int tid   = blockIdx.x * blockDim.x + threadIdx.x;  // 4,096,000 total
    int s     = tid & 15;
    int ln    = (tid >> 4) & 31;
    int c     = (tid >> 9) & 3;
    int ntile = tid >> 11;
    if (ntile >= NTILES) return;
    int k = c * 32 + ((ln >= 16) ? 8 : 0) + ((s < 8) ? s : s + 8);
    int n = ntile * 16 + (ln & 15);
    w2h[tid] = (_Float16)W2[k * VDIM + n];
}

// ---------------------------------------------------------------------------
// Stage 1: h[token][128] = tanh(x @ W1 + b1), stored f16 row-major.
// x = concat(emb[i-2], emb[i-1], emb[i]) gathered through idx into LDS.
// One 128-thread block per token; thread t owns output feature t.
// ---------------------------------------------------------------------------
__global__ void __launch_bounds__(128)
mlp_hidden(const int*   __restrict__ idx,
           const float* __restrict__ wte,
           const float* __restrict__ W1,
           const float* __restrict__ b1,
           _Float16*    __restrict__ hbuf) {
    __shared__ float xs[E3];
    int tok = blockIdx.x;
    int b   = tok >> 10;
    int i   = tok & (TLEN - 1);
    int tid = threadIdx.x;

    for (int k = tid; k < E3; k += 128) {
        int j   = k / EDIM;        // 0,1,2 -> shift 2,1,0
        int e   = k % EDIM;
        int row = i - 2 + j;
        float v = 0.0f;
        if (row >= 0) {
            int w = idx[b * TLEN + row];
            v = wte[w * EDIM + e];
        }
        xs[k] = v;
    }
    __syncthreads();

    float acc = b1[tid];
#pragma unroll 8
    for (int k = 0; k < E3; ++k)
        acc = fmaf(xs[k], W1[k * E2 + tid], acc);   // W1 read coalesced

    hbuf[tok * E2 + tid] = (_Float16)tanhf(acc);
}

// ---------------------------------------------------------------------------
// Stage 2: logits = h @ W2 + b2 via v_wmma_f32_16x16x32_f16.
// Each wave: one 16-row M-tile, 8 consecutive 16-col N-tiles, K=128 (4 chunks).
// A fragment (16-bit 16x32): lane<16 holds row (lane), K {0..7,16..23};
// lane>=16 holds row (lane-16), K {8..15,24..31} -> two aligned 16B loads.
// D layout: lane gives N = lane&15, VGPR r gives M = r + 8*(lane>=16);
// each lane has a fixed output column -> single b2 scalar per tile.
// ---------------------------------------------------------------------------
__global__ void __launch_bounds__(256)
mlp_logits(const _Float16* __restrict__ hbuf,
           const _Float16* __restrict__ w2h,
           const float*    __restrict__ b2,
           float*          __restrict__ out) {
    int lane = threadIdx.x & 31;
    int wave = threadIdx.x >> 5;
    int job  = blockIdx.x * 8 + wave;        // 64000 waves exactly
    int mtile  = job / NGROUPS;
    int ngroup = job % NGROUPS;

    // Load the 4 A chunks for this M-tile once (16 rows x K=128).
    int row  = mtile * 16 + (lane & 15);
    int koff = (lane < 16) ? 0 : 8;
    const _Float16* hrow = hbuf + row * E2 + koff;
    AV a[4];
#pragma unroll
    for (int c = 0; c < 4; ++c) {
        a[c].h[0] = *(const v8h*)(hrow + c * 32);
        a[c].h[1] = *(const v8h*)(hrow + c * 32 + 16);
    }

    for (int t = 0; t < NT_PER_WAVE; ++t) {
        int ntile = ngroup * NT_PER_WAVE + t;
        // B fragments: contiguous 32B per lane per chunk, chunk stride = 32 v16h
        const v16h* bfrag = (const v16h*)w2h + ((size_t)ntile * 4) * 32 + lane;

        v8f acc = {};
#pragma unroll
        for (int c = 0; c < 4; ++c) {
            v16h bv = bfrag[c * 32];
            acc = __builtin_amdgcn_wmma_f32_16x16x32_f16(
                false, a[c].v, false, bv, (short)0, acc, false, false);
        }

        int   col   = ntile * 16 + (lane & 15);
        float bias  = b2[col];
        int   rbase = mtile * 16 + ((lane >= 16) ? 8 : 0);
        size_t obase = (size_t)rbase * VDIM + col;
#pragma unroll
        for (int r = 0; r < 8; ++r)
            out[obase + (size_t)r * VDIM] = acc[r] + bias;
    }
}

extern "C" void kernel_launch(void* const* d_in, const int* in_sizes, int n_in,
                              void* d_out, int out_size, void* d_ws, size_t ws_size,
                              hipStream_t stream) {
    (void)in_sizes; (void)n_in; (void)out_size; (void)ws_size;
    const int*   idx = (const int*)  d_in[0];
    const float* wte = (const float*)d_in[1];
    const float* W1  = (const float*)d_in[2];
    const float* b1  = (const float*)d_in[3];
    const float* W2  = (const float*)d_in[4];
    const float* b2  = (const float*)d_in[5];
    float* out = (float*)d_out;

    // Scratch: [0, 8MB) packed W2 f16 fragments; [8MB, 9MB) h f16 row-major.
    _Float16* w2h  = (_Float16*)d_ws;
    _Float16* hbuf = (_Float16*)((char*)d_ws + (size_t)NTILES * 4 * 32 * 16 * 2);

    pack_w2_f16<<<(NTILES * 4 * 32 * 16 + 255) / 256, 256, 0, stream>>>(W2, w2h);
    mlp_hidden<<<NTOK, 128, 0, stream>>>(idx, wte, W1, b1, hbuf);
    mlp_logits<<<(MTILES * NGROUPS) / 8, 256, 0, stream>>>(hbuf, w2h, b2, out);
}
